// RNN_80479097192732
// MI455X (gfx1250) — compile-verified, matrix operand order checked by reading, not run on hardware
//
#include <hip/hip_runtime.h>

// ---------------------------------------------------------------------------
// RNN (relu cell) for MI455X / gfx1250, bf16 WMMA + async global->LDS path.
//   Z = X @ W_ih^T + b_ih          (one big parallel GEMM, 32768x1024x1024)
//   for t: h = relu(Z_t + h @ W_hh^T + b_hh)   (64 sequential 512x1024x1024)
//   preds = Hs @ fc_w^T + fc_b     (one batched GEMM, 10240x1024x1024)
// GEMM staging: 4-deep async pipeline, 3 K-chunks in flight per wave,
// partial-count s_wait_asynccnt (ASYNCcnt completes in order per ISA).
// ---------------------------------------------------------------------------

typedef __attribute__((ext_vector_type(16))) __bf16 v16bf;
typedef __attribute__((ext_vector_type(8)))  float  v8f;

// Types matching __builtin_amdgcn_global_load_async_to_lds_b128's signature:
// (v4i addrspace(1)* global_src, v4i addrspace(3)* lds_dst, imm off, imm cpol)
typedef int v4i __attribute__((vector_size(16)));
typedef __attribute__((address_space(1))) v4i as1_v4i;
typedef __attribute__((address_space(3))) v4i as3_v4i;

#define HID   1024
#define KDIM  1024
#define TT    64
#define WARM  44
#define TS    20
#define BATCH 512
#define KCH   32            // K-chunk size
#define NCHUNK (KDIM / KCH) // 32 chunks

#if defined(__has_builtin)
#if __has_builtin(__builtin_amdgcn_global_load_async_to_lds_b128) && \
    __has_builtin(__builtin_amdgcn_s_wait_asynccnt)
#define USE_ASYNC_LDS 1
#endif
#endif

#if USE_ASYNC_LDS
#define NBUF 4
#else
#define NBUF 2
#endif

__device__ __forceinline__ unsigned short f2bf(float f) {
    unsigned int u = __builtin_bit_cast(unsigned int, f);
    u += 0x7FFFu + ((u >> 16) & 1u);          // round-to-nearest-even
    return (unsigned short)(u >> 16);
}

__global__ void cvt_f32_bf16(const float* __restrict__ src,
                             unsigned short* __restrict__ dst, int n) {
    for (int i = blockIdx.x * blockDim.x + threadIdx.x; i < n;
         i += gridDim.x * blockDim.x)
        dst[i] = f2bf(src[i]);
}

__global__ void zero_u16(unsigned short* __restrict__ p, int n) {
    for (int i = blockIdx.x * blockDim.x + threadIdx.x; i < n;
         i += gridDim.x * blockDim.x)
        p[i] = 0;
}

union FragU { uint4 q[2]; v16bf v; };

// C[M,N] = A[M,K] @ W[N,K]^T + bias, with mode-specific epilogue.
//   mode 0: Cout[m*HID+n] = v                              (Z gemm)
//   mode 1: v += Z[(m*TT+t)*HID+n]; v = relu(v);
//           HbfOut = bf16(v); if t>=WARM: hs out + Hs_bf   (rnn step)
//   mode 2: Cout[m*HID+n] = v                              (fc gemm)
// Block = 256 threads (8 wave32). Block tile 64(M) x 128(N).
// Wave (wr,wc) tile: 32x32 -> 2x2 WMMA 16x16 f32 accumulators.
__global__ __launch_bounds__(256)
void gemm_wmma(const unsigned short* __restrict__ A,
               const unsigned short* __restrict__ W,
               const float* __restrict__ bias,
               float* __restrict__ Cout,
               const float* __restrict__ Zbase,
               unsigned short* __restrict__ HbfOut,
               unsigned short* __restrict__ HsbfOut,
               int mode, int t)
{
    __shared__ __align__(16) unsigned short As[NBUF][64 * 32];
    __shared__ __align__(16) unsigned short Ws[NBUF][128 * 32];

    const int tid  = threadIdx.x;
    const int lane = tid & 31;
    const int wave = tid >> 5;
    const int wr   = wave >> 2;      // 0..1  (M direction)
    const int wc   = wave & 3;       // 0..3  (N direction)
    const int m0   = blockIdx.y * 64;
    const int n0   = blockIdx.x * 128;
    const int h    = lane >> 4;      // lane half (ISA frag layout)
    const int l15  = lane & 15;

    v8f acc[2][2];
#pragma unroll
    for (int i = 0; i < 2; ++i)
#pragma unroll
        for (int j = 0; j < 2; ++j)
            acc[i][j] = (v8f){0.f, 0.f, 0.f, 0.f, 0.f, 0.f, 0.f, 0.f};

    const int lr = tid >> 2;          // 0..63 : staged row within tile
    const int lk = (tid & 3) * 8;     // 0,8,16,24 : k offset (b128 chunks)

    const unsigned short* ga  = A + (size_t)(m0 + lr)      * KDIM + lk;
    const unsigned short* gw0 = W + (size_t)(n0 + lr)      * KDIM + lk;
    const unsigned short* gw1 = W + (size_t)(n0 + lr + 64) * KDIM + lk;
    const int sOff = lr * 32 + lk;    // LDS element offset for staging

#if USE_ASYNC_LDS
    // Fire-and-forget DMA of one K-chunk slice into LDS buffer `buf`.
    auto stage = [&](int buf, int kc) {
        __builtin_amdgcn_global_load_async_to_lds_b128(
            (as1_v4i*)(ga + kc), (as3_v4i*)&As[buf][sOff], 0, 0);
        __builtin_amdgcn_global_load_async_to_lds_b128(
            (as1_v4i*)(gw0 + kc), (as3_v4i*)&Ws[buf][sOff], 0, 0);
        __builtin_amdgcn_global_load_async_to_lds_b128(
            (as1_v4i*)(gw1 + kc), (as3_v4i*)&Ws[buf][sOff + 64 * 32], 0, 0);
    };
#endif

    auto compute = [&](int buf) {
        FragU af[2], bfrag[2];
#pragma unroll
        for (int i = 0; i < 2; ++i) {
            // A frag: lane holds row (l15), K chunks {8h..8h+7} and {+16}
            int row  = wr * 32 + i * 16 + l15;
            int base = row * 32 + h * 8;
            af[i].q[0] = *(const uint4*)&As[buf][base];
            af[i].q[1] = *(const uint4*)&As[buf][base + 16];
        }
#pragma unroll
        for (int j = 0; j < 2; ++j) {
            // B frag (W row = output col n), mirrored K layout
            int col  = wc * 32 + j * 16 + l15;
            int base = col * 32 + h * 8;
            bfrag[j].q[0] = *(const uint4*)&Ws[buf][base];
            bfrag[j].q[1] = *(const uint4*)&Ws[buf][base + 16];
        }
#pragma unroll
        for (int i = 0; i < 2; ++i)
#pragma unroll
            for (int j = 0; j < 2; ++j)
                acc[i][j] = __builtin_amdgcn_wmma_f32_16x16x32_bf16(
                    false, af[i].v, false, bfrag[j].v,
                    (short)0, acc[i][j], false, false);
    };

#if USE_ASYNC_LDS
    // ---- 4-stage pipeline: keep 3 chunks (9 async loads/wave) in flight ----
    stage(0, 0 * KCH);
    stage(1, 1 * KCH);
    stage(2, 2 * KCH);
    // Iteration ki: wait so chunk ki's 3 loads landed (<=6 newer outstanding),
    // barrier (chunk-ki data visible block-wide; all reads of buffer
    // (ki-1)%4 finished), then stage chunk ki+3 into that buffer, compute ki.
    for (int ki = 0; ki < NCHUNK - 2; ++ki) {
        __builtin_amdgcn_s_wait_asynccnt(6);
        __syncthreads();
        if (ki + 3 < NCHUNK) stage((ki + 3) & 3, (ki + 3) * KCH);
        compute(ki & 3);
    }
    __builtin_amdgcn_s_wait_asynccnt(3);
    __syncthreads();
    compute((NCHUNK - 2) & 3);
    __builtin_amdgcn_s_wait_asynccnt(0);
    __syncthreads();
    compute((NCHUNK - 1) & 3);
#else
    // ---- fallback: synchronous double-buffered staging through VGPRs ----
    {
        uint4 a0 = *(const uint4*)(ga);
        uint4 w0 = *(const uint4*)(gw0);
        uint4 w1 = *(const uint4*)(gw1);
        *(uint4*)&As[0][sOff]           = a0;
        *(uint4*)&Ws[0][sOff]           = w0;
        *(uint4*)&Ws[0][sOff + 64 * 32] = w1;
    }
    __syncthreads();
    for (int kc = 0; kc < KDIM; kc += KCH) {
        const int cur  = (kc >> 5) & 1;
        const int nxt  = cur ^ 1;
        const bool more = (kc + KCH) < KDIM;
        uint4 a0, w0, w1;
        if (more) {
            a0 = *(const uint4*)(ga  + kc + KCH);
            w0 = *(const uint4*)(gw0 + kc + KCH);
            w1 = *(const uint4*)(gw1 + kc + KCH);
        }
        compute(cur);
        if (more) {
            *(uint4*)&As[nxt][sOff]           = a0;
            *(uint4*)&Ws[nxt][sOff]           = w0;
            *(uint4*)&Ws[nxt][sOff + 64 * 32] = w1;
        }
        __syncthreads();
    }
#endif

    // Epilogue: C/D layout -> M = e + 8*h, N = l15 within each 16x16 tile.
#pragma unroll
    for (int i = 0; i < 2; ++i) {
#pragma unroll
        for (int j = 0; j < 2; ++j) {
            const int n  = n0 + wc * 32 + j * 16 + l15;
            const float bn = bias[n];
#pragma unroll
            for (int e = 0; e < 8; ++e) {
                const int m = m0 + wr * 32 + i * 16 + h * 8 + e;
                float v = acc[i][j][e] + bn;
                if (mode == 0) {
                    Cout[(size_t)m * HID + n] = v;
                } else if (mode == 1) {
                    v += Zbase[((size_t)m * TT + t) * HID + n];
                    v = v > 0.f ? v : 0.f;
                    HbfOut[(size_t)m * HID + n] = f2bf(v);
                    if (t >= WARM) {
                        const int row = m * TS + (t - WARM); // [B, TS] layout
                        Cout[(size_t)row * HID + n]    = v;       // hs out (f32)
                        HsbfOut[(size_t)row * HID + n] = f2bf(v); // fc input
                    }
                } else {
                    Cout[(size_t)m * HID + n] = v;  // preds, row = b*TS+s
                }
            }
        }
    }
}

extern "C" void kernel_launch(void* const* d_in, const int* in_sizes, int n_in,
                              void* d_out, int out_size, void* d_ws, size_t ws_size,
                              hipStream_t stream) {
    const float* X    = (const float*)d_in[0];  // [512,64,1024]
    const float* Wih  = (const float*)d_in[1];  // [1024,1024]
    const float* Bih  = (const float*)d_in[2];  // [1024]
    const float* Whh  = (const float*)d_in[3];  // [1024,1024]
    const float* Bhh  = (const float*)d_in[4];  // [1024]
    const float* Fcw  = (const float*)d_in[5];  // [1024,1024]
    const float* Fcb  = (const float*)d_in[6];  // [1024]

    const size_t NX  = (size_t)BATCH * TT * KDIM;   // 33,554,432
    const size_t NW  = (size_t)HID * KDIM;          // 1,048,576
    const size_t NH  = (size_t)BATCH * HID;         // 524,288
    const size_t NHS = (size_t)BATCH * TS * HID;    // 10,485,760

    // Workspace carve-out (256B aligned slices)
    size_t off = 0;
    auto carve = [&](size_t bytes) {
        size_t o = off;
        off = (off + bytes + 255) & ~(size_t)255;
        return o;
    };
    char* ws = (char*)d_ws;
    unsigned short* Xbf   = (unsigned short*)(ws + carve(NX * 2));
    unsigned short* WihB  = (unsigned short*)(ws + carve(NW * 2));
    unsigned short* WhhB  = (unsigned short*)(ws + carve(NW * 2));
    unsigned short* FcwB  = (unsigned short*)(ws + carve(NW * 2));
    float*          Z     = (float*)        (ws + carve(NX * 4));
    unsigned short* Hbf0  = (unsigned short*)(ws + carve(NH * 2));
    unsigned short* Hbf1  = (unsigned short*)(ws + carve(NH * 2));
    unsigned short* Hsbf  = (unsigned short*)(ws + carve(NHS * 2));
    (void)ws_size; (void)in_sizes; (void)n_in; (void)out_size;

    float* preds = (float*)d_out;                               // [512,20,1024]
    float* hsout = (float*)d_out + (size_t)BATCH * TS * HID;    // [512,20,1024]

    // 1) convert inputs / weights to bf16
    cvt_f32_bf16<<<dim3(4096), dim3(256), 0, stream>>>(X,   Xbf,  (int)NX);
    cvt_f32_bf16<<<dim3(1024), dim3(256), 0, stream>>>(Wih, WihB, (int)NW);
    cvt_f32_bf16<<<dim3(1024), dim3(256), 0, stream>>>(Whh, WhhB, (int)NW);
    cvt_f32_bf16<<<dim3(1024), dim3(256), 0, stream>>>(Fcw, FcwB, (int)NW);
    zero_u16   <<<dim3(512),  dim3(256), 0, stream>>>(Hbf0, (int)NH);

    // 2) Z = X @ Wih^T + b_ih   (M = 32768 -> 512 M-tiles, 8 N-tiles)
    gemm_wmma<<<dim3(8, 512), dim3(256), 0, stream>>>(
        Xbf, WihB, Bih, Z, nullptr, nullptr, nullptr, /*mode=*/0, 0);

    // 3) recurrence: h = relu(Z_t + h @ Whh^T + b_hh), ping-pong bf16 h
    for (int t = 0; t < TT; ++t) {
        unsigned short* hin  = (t & 1) ? Hbf1 : Hbf0;
        unsigned short* hout = (t & 1) ? Hbf0 : Hbf1;
        gemm_wmma<<<dim3(8, 8), dim3(256), 0, stream>>>(
            hin, WhhB, Bhh, hsout, Z, hout, Hsbf, /*mode=*/1, t);
    }

    // 4) preds = Hs @ fc_w^T + fc_b   (M = 10240 -> 160 M-tiles)
    gemm_wmma<<<dim3(8, 160), dim3(256), 0, stream>>>(
        Hsbf, FcwB, Fcb, preds, nullptr, nullptr, nullptr, /*mode=*/2, 0);
}
